// Block_34711925686730
// MI455X (gfx1250) — matrix operand order repaired and checked by reading, not compiled, not run on hardware
//
#include <hip/hip_runtime.h>
#include <math.h>

// ---------------------------------------------------------------------------
// Problem constants
// ---------------------------------------------------------------------------
#define DIMC   384
#define HEADS  6
#define GCH    64          // channels per group
#define KPATCH 576         // 64 * 3 * 3 (implicit-GEMM K for grouped conv)
#define NTOK   25088       // 32 * 28 * 28
#define HW     784         // 28*28
#define ITERS  25
#define MLPH   1536
#define EPSV   1e-6f
#define APAD   584         // 576 + 8 bf16 pad -> conflict-free b128 LDS reads
#define GPAD   392         // 384 + 8 bf16 pad
#define MT     32          // tokens per block tile (2 x 16 WMMA M-tiles)

typedef __bf16 bf16_t;
typedef __attribute__((ext_vector_type(16))) __bf16 v16bf;
typedef __attribute__((ext_vector_type(8)))  float  v8f;

union FragAB { v16bf v; uint4 q[2]; };

__device__ __forceinline__ float bf2f(unsigned short u) {
    return __uint_as_float(((unsigned int)u) << 16);
}
__device__ __forceinline__ unsigned short f2bf(float f) {
    unsigned int x = __float_as_uint(f);
    x += 0x7FFFu + ((x >> 16) & 1u);          // round-to-nearest-even
    return (unsigned short)(x >> 16);
}
// v_rcp_f32 + one Newton-Raphson step (plenty for NNMF renormalized updates)
__device__ __forceinline__ float fastdiv(float a, float b) {
    float r = __builtin_amdgcn_rcpf(b);
    r = r * (2.0f - b * r);
    return a * r;
}

// ---------------------------------------------------------------------------
// Weight prep: wp = |Wc| L1-normalized per out-channel.
//   Wf[g][co][ (ky*3+kx)*64 + ci ] = wp           (forward conv GEMM B, K-major)
//   Wb[g][ci][ (8-(ky*3+kx))*64 + co ] = wp       (transposed/flipped kernel)
// ---------------------------------------------------------------------------
__global__ __launch_bounds__(128) void wprep_kernel(
    const float* __restrict__ Wc, unsigned short* __restrict__ Wf,
    unsigned short* __restrict__ Wb)
{
    __shared__ float red[128];
    const int co = blockIdx.x;            // 0..383
    const int g = co / GCH, coi = co % GCH;
    const int tid = threadIdx.x;

    float s = 0.f;
    for (int k = tid; k < KPATCH; k += 128) s += fabsf(Wc[(size_t)co * KPATCH + k]);
    red[tid] = s;
    __syncthreads();
    for (int off = 64; off > 0; off >>= 1) {
        if (tid < off) red[tid] += red[tid + off];
        __syncthreads();
    }
    const float inv = fastdiv(1.f, fmaxf(red[0], EPSV));

    for (int k = tid; k < KPATCH; k += 128) {
        float a = fabsf(Wc[(size_t)co * KPATCH + k]) * inv;
        unsigned short b = f2bf(a);
        int ci = k / 9, r = k % 9;        // r = ky*3+kx
        Wf[((size_t)(g * GCH + coi)) * KPATCH + r * GCH + ci] = b;
        Wb[((size_t)(g * GCH + ci)) * KPATCH + (8 - r) * GCH + coi] = b;
    }
}

// Generic fp32 [K][N] -> bf16 [N][K] transpose (for MLP weights)
__global__ void transpose_to_bf16(const float* __restrict__ in,
                                  unsigned short* __restrict__ out, int K, int N)
{
    int idx = blockIdx.x * 256 + threadIdx.x;
    if (idx >= K * N) return;
    int n = idx / K, k = idx % K;
    out[idx] = f2bf(in[(size_t)k * N + n]);
}

// ---------------------------------------------------------------------------
// Input prep: channels-first LayerNorm + clamp(eps) + channel-sum normalize.
// One wave32 per token. Also stores x in NHWC fp32 and h = 1/384.
// ---------------------------------------------------------------------------
__global__ __launch_bounds__(256) void inprep_kernel(
    const float* __restrict__ x, const float* __restrict__ g1,
    const float* __restrict__ b1, unsigned short* __restrict__ xn,
    float* __restrict__ xr, float* __restrict__ h, unsigned short* __restrict__ hbf)
{
    const int lane = threadIdx.x & 31, wv = threadIdx.x >> 5;
    const int token = blockIdx.x * 8 + wv;
    const int n = token / HW, rem = token % HW;
    const float* base = x + (size_t)n * DIMC * HW + rem;

    float v[12], s = 0.f, ss = 0.f;
    #pragma unroll
    for (int i = 0; i < 12; ++i) {
        int c = lane + 32 * i;
        v[i] = base[(size_t)c * HW];
        s += v[i]; ss += v[i] * v[i];
    }
    #pragma unroll
    for (int off = 16; off > 0; off >>= 1) { s += __shfl_xor(s, off); ss += __shfl_xor(ss, off); }
    const float mu = s * (1.f / DIMC);
    const float var = ss * (1.f / DIMC) - mu * mu;
    const float rs = rsqrtf(var + 1e-6f);

    float w[12]; float s2 = 0.f;
    #pragma unroll
    for (int i = 0; i < 12; ++i) {
        int c = lane + 32 * i;
        w[i] = fmaxf((v[i] - mu) * rs * g1[c] + b1[c], EPSV);
        s2 += w[i];
    }
    #pragma unroll
    for (int off = 16; off > 0; off >>= 1) s2 += __shfl_xor(s2, off);
    const float inv = fastdiv(1.f, fmaxf(s2, EPSV));
    const float hinit = 1.f / DIMC;
    const unsigned short hinit_b = f2bf(hinit);

    #pragma unroll
    for (int i = 0; i < 12; ++i) {
        int c = lane + 32 * i;
        size_t idx = (size_t)token * DIMC + c;
        xn[idx]  = f2bf(w[i] * inv);
        xr[idx]  = v[i];
        h[idx]   = hinit;
        hbf[idx] = hinit_b;
    }
}

// ---------------------------------------------------------------------------
// Implicit-GEMM grouped conv core: 32 tokens x 64 out-ch per block,
// each wave does a 32x16 output (two 16x16 WMMA M-tiles sharing B fragments).
// ---------------------------------------------------------------------------
__device__ __forceinline__ void stage_patch(unsigned short* Alds,
    const unsigned short* __restrict__ act, int ttile, int g, int tid)
{
    for (int s = tid; s < MT * 9 * 4; s += 128) {   // 32 tok * 9 pos * 4x16 bf16
        int seg = s >> 2, part = s & 3;
        int ti = seg / 9, pos = seg % 9;
        int ky = pos / 3, kx = pos % 3;
        int token = ttile + ti;
        int n = token / HW, rem = token % HW;
        int y = rem / 28, xx = rem % 28;
        int yy = y + ky - 1, xs = xx + kx - 1;
        uint4 val = make_uint4(0u, 0u, 0u, 0u);
        if (yy >= 0 && yy < 28 && xs >= 0 && xs < 28) {
            val = *(const uint4*)(act + ((size_t)(n * HW + yy * 28 + xs)) * DIMC
                                      + g * GCH + part * 16);
        }
        *(uint4*)(Alds + ti * APAD + pos * GCH + part * 16) = val;
    }
}

__device__ __forceinline__ void conv_mma2(const unsigned short* Alds,
    const unsigned short* __restrict__ Wg, int lane, int nOff,
    v8f& acc0, v8f& acc1)
{
    const int lm = lane & 15, lh = lane >> 4;
    const unsigned short* wrow  = Wg + (size_t)(nOff + lm) * KPATCH;
    const unsigned short* arow0 = Alds + lm * APAD;
    const unsigned short* arow1 = Alds + (16 + lm) * APAD;
    #pragma unroll 2
    for (int kk = 0; kk < 18; ++kk) {
        FragAB a0, a1, b;
        const int ka = kk * 32 + lh * 8;
        const int kb = kk * 32 + lh * 8;
        b.q[0]  = *(const uint4*)(wrow + kb);
        b.q[1]  = *(const uint4*)(wrow + kb + 16);
        a0.q[0] = *(const uint4*)(arow0 + ka);
        a0.q[1] = *(const uint4*)(arow0 + ka + 16);
        a1.q[0] = *(const uint4*)(arow1 + ka);
        a1.q[1] = *(const uint4*)(arow1 + ka + 16);
        acc0 = __builtin_amdgcn_wmma_f32_16x16x32_bf16(false, a0.v, false, b.v,
                                                       (short)0, acc0, false, false);
        acc1 = __builtin_amdgcn_wmma_f32_16x16x32_bf16(false, a1.v, false, b.v,
                                                       (short)0, acc1, false, false);
    }
}

// recon = conv(h, wt) fused with nu = xn / max(recon, eps)
__global__ __launch_bounds__(128) void conv_nu_kernel(
    const unsigned short* __restrict__ hbf, const unsigned short* __restrict__ Wb,
    const unsigned short* __restrict__ xn, unsigned short* __restrict__ nu)
{
    __shared__ __align__(16) unsigned short Alds[MT * APAD];
    const int tid = threadIdx.x, lane = tid & 31, wv = tid >> 5;
    const int ttile = blockIdx.x * MT, g = blockIdx.y;

    stage_patch(Alds, hbf, ttile, g, tid);
    __syncthreads();

    v8f acc0 = {}, acc1 = {};
    conv_mma2(Alds, Wb + (size_t)g * GCH * KPATCH, lane, wv * 16, acc0, acc1);

    const int lm = lane & 15, lh = lane >> 4;
    const int c = g * GCH + wv * 16 + lm;
    #pragma unroll
    for (int t = 0; t < 2; ++t) {
        const v8f& acc = t ? acc1 : acc0;
        #pragma unroll
        for (int j = 0; j < 8; ++j) {
            int token = ttile + t * 16 + lh * 8 + j;
            size_t idx = (size_t)token * DIMC + c;
            nu[idx] = f2bf(fastdiv(bf2f(xn[idx]), fmaxf(acc[j], EPSV)));
        }
    }
}

// hr = h * conv(nu, wp)    (pre-normalization multiplicative update)
__global__ __launch_bounds__(128) void conv_upd_kernel(
    const unsigned short* __restrict__ nu, const unsigned short* __restrict__ Wf,
    const float* __restrict__ h, float* __restrict__ hr)
{
    __shared__ __align__(16) unsigned short Alds[MT * APAD];
    const int tid = threadIdx.x, lane = tid & 31, wv = tid >> 5;
    const int ttile = blockIdx.x * MT, g = blockIdx.y;

    stage_patch(Alds, nu, ttile, g, tid);
    __syncthreads();

    v8f acc0 = {}, acc1 = {};
    conv_mma2(Alds, Wf + (size_t)g * GCH * KPATCH, lane, wv * 16, acc0, acc1);

    const int lm = lane & 15, lh = lane >> 4;
    const int c = g * GCH + wv * 16 + lm;
    #pragma unroll
    for (int t = 0; t < 2; ++t) {
        const v8f& acc = t ? acc1 : acc0;
        #pragma unroll
        for (int j = 0; j < 8; ++j) {
            int token = ttile + t * 16 + lh * 8 + j;
            size_t idx = (size_t)token * DIMC + c;
            hr[idx] = h[idx] * acc[j];
        }
    }
}

// h = hr / max(sum_c hr, eps); refresh fp32 + bf16 copies. Wave per token.
__global__ __launch_bounds__(256) void norm_kernel(
    const float* __restrict__ hr, float* __restrict__ h,
    unsigned short* __restrict__ hbf)
{
    const int lane = threadIdx.x & 31, wv = threadIdx.x >> 5;
    const int token = blockIdx.x * 8 + wv;
    const float* row = hr + (size_t)token * DIMC;

    float v[12], s = 0.f;
    #pragma unroll
    for (int i = 0; i < 12; ++i) { v[i] = row[lane + 32 * i]; s += v[i]; }
    #pragma unroll
    for (int off = 16; off > 0; off >>= 1) s += __shfl_xor(s, off);
    const float inv = fastdiv(1.f, fmaxf(s, EPSV));

    #pragma unroll
    for (int i = 0; i < 12; ++i) {
        size_t idx = (size_t)token * DIMC + lane + 32 * i;
        float hv = v[i] * inv;
        h[idx] = hv;
        hbf[idx] = f2bf(hv);
    }
}

// xr += h (attn residual); yn = LayerNorm_last(xr) * g2 + b2  (eps=1e-5)
__global__ __launch_bounds__(256) void res_ln_kernel(
    float* __restrict__ xr, const float* __restrict__ h,
    const float* __restrict__ g2, const float* __restrict__ b2,
    unsigned short* __restrict__ yn)
{
    const int lane = threadIdx.x & 31, wv = threadIdx.x >> 5;
    const int token = blockIdx.x * 8 + wv;

    float v[12], s = 0.f, ss = 0.f;
    #pragma unroll
    for (int i = 0; i < 12; ++i) {
        size_t idx = (size_t)token * DIMC + lane + 32 * i;
        v[i] = xr[idx] + h[idx];
        xr[idx] = v[i];
        s += v[i]; ss += v[i] * v[i];
    }
    #pragma unroll
    for (int off = 16; off > 0; off >>= 1) { s += __shfl_xor(s, off); ss += __shfl_xor(ss, off); }
    const float mu = s * (1.f / DIMC);
    const float var = ss * (1.f / DIMC) - mu * mu;
    const float rs = rsqrtf(var + 1e-5f);

    #pragma unroll
    for (int i = 0; i < 12; ++i) {
        int c = lane + 32 * i;
        yn[(size_t)token * DIMC + c] = f2bf((v[i] - mu) * rs * g2[c] + b2[c]);
    }
}

// ---------------------------------------------------------------------------
// MLP GEMM1: hid = gelu(yn @ w_fc1 + b_fc1), K=384, tile 32 tok x 64 col
// ---------------------------------------------------------------------------
__global__ __launch_bounds__(128) void gemm1_kernel(
    const unsigned short* __restrict__ yn, const unsigned short* __restrict__ w1t,
    const float* __restrict__ bfc1, unsigned short* __restrict__ hid)
{
    __shared__ __align__(16) unsigned short Alds[MT * GPAD];
    const int tid = threadIdx.x, lane = tid & 31, wv = tid >> 5;
    const int ttile = blockIdx.x * MT, ncol = blockIdx.y * 64;

    for (int s = tid; s < MT * 48; s += 128) {             // 32 rows * 48 uint4
        int ti = s / 48, part = s % 48;
        *(uint4*)(Alds + ti * GPAD + part * 8) =
            *(const uint4*)(yn + (size_t)(ttile + ti) * DIMC + part * 8);
    }
    __syncthreads();

    const int lm = lane & 15, lh = lane >> 4;
    const int nOff = ncol + wv * 16;
    const unsigned short* wrow  = w1t + (size_t)(nOff + lm) * DIMC;
    const unsigned short* arow0 = Alds + lm * GPAD;
    const unsigned short* arow1 = Alds + (16 + lm) * GPAD;

    v8f acc0 = {}, acc1 = {};
    #pragma unroll
    for (int kk = 0; kk < 12; ++kk) {
        FragAB a0, a1, b;
        const int ka = kk * 32 + lh * 8;
        const int kb = kk * 32 + lh * 8;
        b.q[0]  = *(const uint4*)(wrow + kb);
        b.q[1]  = *(const uint4*)(wrow + kb + 16);
        a0.q[0] = *(const uint4*)(arow0 + ka);
        a0.q[1] = *(const uint4*)(arow0 + ka + 16);
        a1.q[0] = *(const uint4*)(arow1 + ka);
        a1.q[1] = *(const uint4*)(arow1 + ka + 16);
        acc0 = __builtin_amdgcn_wmma_f32_16x16x32_bf16(false, a0.v, false, b.v,
                                                       (short)0, acc0, false, false);
        acc1 = __builtin_amdgcn_wmma_f32_16x16x32_bf16(false, a1.v, false, b.v,
                                                       (short)0, acc1, false, false);
    }

    const int col = nOff + lm;
    const float bb = bfc1[col];
    #pragma unroll
    for (int t = 0; t < 2; ++t) {
        const v8f& acc = t ? acc1 : acc0;
        #pragma unroll
        for (int j = 0; j < 8; ++j) {
            int token = ttile + t * 16 + lh * 8 + j;
            float gv = acc[j] + bb;
            float ge = 0.5f * gv * (1.f + erff(gv * 0.70710678f));   // exact gelu
            hid[(size_t)token * MLPH + col] = f2bf(ge);
        }
    }
}

// MLP GEMM2: out = xr + hid @ w_fc2 + b_fc2, K=1536 (staged in 4 chunks),
// stores fp32 NCHW directly into d_out.
__global__ __launch_bounds__(128) void gemm2_kernel(
    const unsigned short* __restrict__ hid, const unsigned short* __restrict__ w2t,
    const float* __restrict__ bfc2, const float* __restrict__ xr,
    float* __restrict__ out)
{
    __shared__ __align__(16) unsigned short Alds[MT * GPAD];
    const int tid = threadIdx.x, lane = tid & 31, wv = tid >> 5;
    const int ttile = blockIdx.x * MT, ncol = blockIdx.y * 64;
    const int lm = lane & 15, lh = lane >> 4;
    const int nOff = ncol + wv * 16;

    v8f acc0 = {}, acc1 = {};
    for (int ch = 0; ch < 4; ++ch) {
        __syncthreads();
        for (int s = tid; s < MT * 48; s += 128) {
            int ti = s / 48, part = s % 48;
            *(uint4*)(Alds + ti * GPAD + part * 8) =
                *(const uint4*)(hid + (size_t)(ttile + ti) * MLPH + ch * DIMC + part * 8);
        }
        __syncthreads();

        const unsigned short* wrow  = w2t + (size_t)(nOff + lm) * MLPH + ch * DIMC;
        const unsigned short* arow0 = Alds + lm * GPAD;
        const unsigned short* arow1 = Alds + (16 + lm) * GPAD;
        #pragma unroll
        for (int kk = 0; kk < 12; ++kk) {
            FragAB a0, a1, b;
            const int ka = kk * 32 + lh * 8;
            const int kb = kk * 32 + lh * 8;
            b.q[0]  = *(const uint4*)(wrow + kb);
            b.q[1]  = *(const uint4*)(wrow + kb + 16);
            a0.q[0] = *(const uint4*)(arow0 + ka);
            a0.q[1] = *(const uint4*)(arow0 + ka + 16);
            a1.q[0] = *(const uint4*)(arow1 + ka);
            a1.q[1] = *(const uint4*)(arow1 + ka + 16);
            acc0 = __builtin_amdgcn_wmma_f32_16x16x32_bf16(false, a0.v, false, b.v,
                                                           (short)0, acc0, false, false);
            acc1 = __builtin_amdgcn_wmma_f32_16x16x32_bf16(false, a1.v, false, b.v,
                                                           (short)0, acc1, false, false);
        }
    }

    const int col = nOff + lm;
    const float bb = bfc2[col];
    #pragma unroll
    for (int t = 0; t < 2; ++t) {
        const v8f& acc = t ? acc1 : acc0;
        #pragma unroll
        for (int j = 0; j < 8; ++j) {
            int token = ttile + t * 16 + lh * 8 + j;
            float val = acc[j] + bb + xr[(size_t)token * DIMC + col];
            int n = token / HW, rem = token % HW;
            out[((size_t)n * DIMC + col) * HW + rem] = val;   // NCHW
        }
    }
}

// ---------------------------------------------------------------------------
// Host launcher
// ---------------------------------------------------------------------------
extern "C" void kernel_launch(void* const* d_in, const int* in_sizes, int n_in,
                              void* d_out, int out_size, void* d_ws, size_t ws_size,
                              hipStream_t stream)
{
    const float* x     = (const float*)d_in[0];
    const float* g1    = (const float*)d_in[1];
    const float* b1    = (const float*)d_in[2];
    const float* Wc    = (const float*)d_in[3];
    const float* g2    = (const float*)d_in[4];
    const float* b2    = (const float*)d_in[5];
    const float* w_fc1 = (const float*)d_in[6];
    const float* b_fc1 = (const float*)d_in[7];
    const float* w_fc2 = (const float*)d_in[8];
    const float* b_fc2 = (const float*)d_in[9];
    float* out = (float*)d_out;

    const size_t SZ_F32 = (size_t)NTOK * DIMC * 4;      // 38,535,168
    const size_t SZ_B16 = (size_t)NTOK * DIMC * 2;      // 19,267,584
    const size_t SZ_W   = (size_t)HEADS * GCH * KPATCH * 2;
    const size_t SZ_W1  = (size_t)DIMC * MLPH * 2;

    char* w = (char*)d_ws;
    size_t o = 0;
    float*          XR  = (float*)(w + o);          o += SZ_F32;
    float*          H   = (float*)(w + o);          o += SZ_F32;
    unsigned short* XN  = (unsigned short*)(w + o); o += SZ_B16;
    unsigned short* HBF = (unsigned short*)(w + o); o += SZ_B16;  // |
    unsigned short* NU  = (unsigned short*)(w + o); o += SZ_B16;  // | contiguous ->
    float*          HR  = (float*)(w + o);          o += SZ_F32;  // | reused as HID
    unsigned short* YN  = (unsigned short*)(w + o); o += SZ_B16;
    unsigned short* WF  = (unsigned short*)(w + o); o += SZ_W;
    unsigned short* WB  = (unsigned short*)(w + o); o += SZ_W;
    unsigned short* W1T = (unsigned short*)(w + o); o += SZ_W1;
    unsigned short* W2T = (unsigned short*)(w + o); o += SZ_W1;
    unsigned short* HID = HBF;                       // 2*SZ_B16 + SZ_F32 == NTOK*MLPH*2

    // --- one-time prep ---
    wprep_kernel<<<dim3(DIMC), dim3(128), 0, stream>>>(Wc, WF, WB);
    transpose_to_bf16<<<dim3((DIMC * MLPH + 255) / 256), dim3(256), 0, stream>>>(
        w_fc1, W1T, DIMC, MLPH);
    transpose_to_bf16<<<dim3((DIMC * MLPH + 255) / 256), dim3(256), 0, stream>>>(
        w_fc2, W2T, MLPH, DIMC);
    inprep_kernel<<<dim3(NTOK / 8), dim3(256), 0, stream>>>(x, g1, b1, XN, XR, H, HBF);

    // --- NNMF multiplicative updates ---
    const dim3 cgrid(NTOK / MT, HEADS);
    for (int it = 0; it < ITERS; ++it) {
        conv_nu_kernel <<<cgrid, dim3(128), 0, stream>>>(HBF, WB, XN, NU);
        conv_upd_kernel<<<cgrid, dim3(128), 0, stream>>>(NU, WF, H, HR);
        norm_kernel    <<<dim3(NTOK / 8), dim3(256), 0, stream>>>(HR, H, HBF);
    }

    // --- residual + LN + MLP ---
    res_ln_kernel<<<dim3(NTOK / 8), dim3(256), 0, stream>>>(XR, H, g2, b2, YN);
    gemm1_kernel <<<dim3(NTOK / MT, MLPH / 64), dim3(128), 0, stream>>>(YN, W1T, b_fc1, HID);
    gemm2_kernel <<<dim3(NTOK / MT, DIMC / 64), dim3(128), 0, stream>>>(HID, W2T, b_fc2, XR, out);
}
